// SOPDilateLoss_31353261261224
// MI455X (gfx1250) — matrix operand order repaired and checked by reading, not compiled, not run on hardware
//
#include <hip/hip_runtime.h>
#include <hip/hip_bf16.h>

typedef __attribute__((ext_vector_type(16))) _Float16 v16h;
typedef __attribute__((ext_vector_type(8)))  _Float16 v8h;
typedef __attribute__((ext_vector_type(8)))  float    v8f;

#define SDTW_INF 1e10f
#define SDTW_GAMMA 0.1f
#define MAX_DIAG 2052      // supports m up to 2050
#define MAX_SEG 16
#define DTW_THREADS 1024
#define SLOTS 3            // ceil(2050 / 1024)

__device__ __forceinline__ int imax(int a, int b) { return a > b ? a : b; }
__device__ __forceinline__ int imin(int a, int b) { return a < b ? a : b; }

// ---------------------------------------------------------------------------
// Diagonal-packed layout: cell (i,j), 1-based, lives at off[d] + i - i_lo(d),
// d = i + j, i_lo(d) = max(1, d - n). Total cells = m*n (tight packing).
// off[d] = #cells with i+j < d.
// ---------------------------------------------------------------------------
__global__ void build_offsets(int* __restrict__ off, int m, int n) {
    if (blockIdx.x == 0 && threadIdx.x == 0) {
        int acc = 0;
        for (int d = 0; d <= m + n + 1; ++d) {
            off[d] = acc;
            int lo = imax(1, d - n), hi = imin(m, d - 1);
            int len = hi - lo + 1;
            if (len > 0) acc += len;
        }
    }
}

// ---------------------------------------------------------------------------
__global__ void cvt_f32_to_f16(const float* __restrict__ src, _Float16* __restrict__ dst, int n) {
    int i = blockIdx.x * blockDim.x + threadIdx.x;
    if (i < n) dst[i] = (_Float16)src[i];
}

__global__ void row_norms(const float* __restrict__ src, float* __restrict__ out, int rows, int D) {
    int wave = threadIdx.x >> 5;
    int lane = threadIdx.x & 31;
    int row  = blockIdx.x * (blockDim.x >> 5) + wave;
    if (row >= rows) return;
    float s = 0.f;
    for (int k = lane; k < D; k += 32) {
        float v = src[(size_t)row * D + k];
        s += v * v;
    }
    #pragma unroll
    for (int off = 16; off > 0; off >>= 1) s += __shfl_xor(s, off);
    if (lane == 0) out[row] = s;
}

__global__ void zero_stats(float* __restrict__ stats) {
    if (threadIdx.x < 64) stats[threadIdx.x] = 0.f;
}

// ---------------------------------------------------------------------------
// Cost via WMMA, epilogue scatters into diagonal-packed Cd.
// Cd[off[d] + i1 - i_lo(d)] = max(t2 + p2 - 2*dot, 0),  d = i1 + j1 (1-based)
// ---------------------------------------------------------------------------
__global__ __launch_bounds__(256) void cost_wmma(
    const _Float16* __restrict__ Tg, const _Float16* __restrict__ Pd,
    const float* __restrict__ t2, const float* __restrict__ p2,
    const int* __restrict__ off,
    float* __restrict__ Cd, int M, int N) {

    int wave = threadIdx.x >> 5;
    int lane = threadIdx.x & 31;
    int tilesPerRow = N >> 4;
    int tileId = blockIdx.x * 8 + wave;
    if (tileId >= (M >> 4) * tilesPerRow) return;
    int tm = tileId / tilesPerRow;
    int tn = tileId % tilesPerRow;

    int r = lane & 15;
    int h = lane >> 4;

    // A fragments (16-bit A 16x32 layout)
    const _Float16* arow = Tg + (size_t)(tm * 16 + r) * 64;
    v8h a00 = *(const v8h*)(arow +      8 * h);
    v8h a01 = *(const v8h*)(arow + 16 + 8 * h);
    v8h a10 = *(const v8h*)(arow + 32 + 8 * h);
    v8h a11 = *(const v8h*)(arow + 48 + 8 * h);
    v16h a0, a1;
    #pragma unroll
    for (int e = 0; e < 8; ++e) {
        a0[e] = a00[e]; a0[e + 8] = a01[e];
        a1[e] = a10[e]; a1[e + 8] = a11[e];
    }

    // B fragments (16-bit B 32x16 layout)
    const _Float16* brow = Pd + (size_t)(tn * 16 + r) * 64;
    v8h b00 = *(const v8h*)(brow +      8 * h);
    v8h b01 = *(const v8h*)(brow + 16 + 8 * h);
    v8h b10 = *(const v8h*)(brow + 32 + 8 * h);
    v8h b11 = *(const v8h*)(brow + 48 + 8 * h);
    v16h b0, b1;
    #pragma unroll
    for (int e = 0; e < 8; ++e) {
        b0[e] = b00[e]; b0[e + 8] = b01[e];
        b1[e] = b10[e]; b1[e + 8] = b11[e];
    }

    v8f acc = {};
    acc = __builtin_amdgcn_wmma_f32_16x16x32_f16(false, a0, false, b0, (short)0, acc, false, false);
    acc = __builtin_amdgcn_wmma_f32_16x16x32_f16(false, a1, false, b1, (short)0, acc, false, false);

    int j0 = tn * 16 + r;            // 0-based column
    float p2j = p2[j0];
    #pragma unroll
    for (int v = 0; v < 8; ++v) {
        int i0 = tm * 16 + v + 8 * h;    // 0-based row
        float c = fmaxf(t2[i0] + p2j - 2.0f * acc[v], 0.0f);
        int d = i0 + j0 + 2;
        int ilo = imax(1, d - N);
        Cd[(size_t)off[d] + (i0 + 1 - ilo)] = c;
    }
}

// ---------------------------------------------------------------------------
__device__ __forceinline__ float softmin3(float a, float b, float c, float inv_g, float g) {
    float mn = fminf(a, fminf(b, c));
    float s = __expf((mn - a) * inv_g) + __expf((mn - b) * inv_g) + __expf((mn - c) * inv_g);
    return mn - g * __logf(s);
}

// ---------------------------------------------------------------------------
// Fused soft-DTW: block 0 = forward, block 1 = backward (independent given Cd;
// run concurrently on two WGPs). Diagonal-packed, coalesced, software-pipelined
// cost prefetch. Fd/Bd stored diagonal-packed; Fd[off[m+n]] = R_fwd[m][n].
// ---------------------------------------------------------------------------
__global__ __launch_bounds__(DTW_THREADS) void dtw_both(
    const float* __restrict__ Cd, float* __restrict__ Fd, float* __restrict__ Bd,
    const int* __restrict__ off, int m, int n, float g) {

    __shared__ float buf[3][MAX_DIAG];   // R diagonals
    __shared__ float cb[3][MAX_DIAG];    // cost diagonals (backward only)
    const float inv_g = 1.0f / g;
    int tid = threadIdx.x;

    if (blockIdx.x == 0) {
        // ---------------- forward ----------------
        for (int i = tid; i <= m; i += DTW_THREADS) {
            buf[0][i] = (i == 0) ? 0.0f : SDTW_INF;   // diagonal d=0
            buf[1][i] = SDTW_INF;                      // diagonal d=1
        }
        float cCur[SLOTS];
        {   // preload cost diagonal d=2
            int ilo = imax(1, 2 - n), ihi = imin(m, 1);
            int base = off[2] - ilo;
            #pragma unroll
            for (int s = 0; s < SLOTS; ++s) {
                int i = tid + s * DTW_THREADS;
                cCur[s] = (i >= ilo && i <= ihi) ? Cd[base + i] : 0.0f;
            }
        }
        __syncthreads();

        int pm2 = 0, pm1 = 1, pnw = 2;
        for (int d = 2; d <= m + n; ++d) {
            float* Dm2 = buf[pm2];
            float* Dm1 = buf[pm1];
            float* Dnw = buf[pnw];
            int ilo = imax(1, d - n), ihi = imin(m, d - 1);
            int base = off[d] - ilo;

            // prefetch next diagonal's cost (loads issue before compute)
            float cNxt[SLOTS];
            int d1 = d + 1;
            int ilo1 = imax(1, d1 - n);
            int ihi1 = (d1 <= m + n) ? imin(m, d1 - 1) : 0;
            int base1 = (d1 <= m + n) ? (off[d1] - ilo1) : 0;
            #pragma unroll
            for (int s = 0; s < SLOTS; ++s) {
                int i = tid + s * DTW_THREADS;
                cNxt[s] = (i >= ilo1 && i <= ihi1) ? Cd[base1 + i] : 0.0f;
            }

            #pragma unroll
            for (int s = 0; s < SLOTS; ++s) {
                int i = tid + s * DTW_THREADS;
                float val = SDTW_INF;
                if (i >= ilo && i <= ihi) {
                    float sm = softmin3(Dm2[i - 1], Dm1[i - 1], Dm1[i], inv_g, g);
                    val = cCur[s] + sm;
                    Fd[base + i] = val;
                }
                if (i <= m) Dnw[i] = val;
            }
            __syncthreads();
            #pragma unroll
            for (int s = 0; s < SLOTS; ++s) cCur[s] = cNxt[s];
            int t = pm2; pm2 = pm1; pm1 = pnw; pnw = t;
        }
    } else {
        // ---------------- backward ----------------
        for (int k = tid; k <= m + 1; k += DTW_THREADS) {
            buf[0][k] = (k == m) ? 0.0f : SDTW_INF;   // diagonal d=m+n (R_bwd[m][n]=0)
            buf[1][k] = SDTW_INF;                      // diagonal d=m+n+1
            cb[0][k] = 0.0f;                           // cost diag m+n (only row m valid)
            cb[1][k] = 0.0f;                           // cost diag m+n+1 (never read)
        }
        if (tid == 0) {
            cb[0][m] = Cd[off[m + n]];                 // cost cell (m,n)
            Bd[off[m + n]] = 0.0f;                     // R_bwd cell (m,n) = 0
        }
        __syncthreads();

        int pe1 = 0, pe2 = 1, pen = 2;   // R_bwd diagonals d+1, d+2, new
        int pc1 = 0, pc2 = 1, pcn = 2;   // cost diagonals d+1, d+2, new(d)
        for (int d = m + n - 1; d >= 1; --d) {
            float* Ep1 = buf[pe1];
            float* Ep2 = buf[pe2];
            float* Enw = buf[pen];
            float* C1  = cb[pc1];
            float* C2  = cb[pc2];
            float* Cn  = cb[pcn];
            int ilo = imax(1, d - n), ihi = imin(m, d - 1);
            int base = off[d] - ilo;

            // issue loads for cost diagonal d (consumed next iteration)
            float cl[SLOTS];
            #pragma unroll
            for (int s = 0; s < SLOTS; ++s) {
                int r = tid + s * DTW_THREADS;
                cl[s] = (r >= ilo && r <= ihi) ? Cd[base + r] : 0.0f;
            }

            #pragma unroll
            for (int s = 0; s < SLOTS; ++s) {
                int k = tid + s * DTW_THREADS;
                int j = d - k;
                float e = SDTW_INF;
                if (k >= 1 && k <= m && j >= 1 && j <= n) {
                    // cost cell (k+1,j+1) -> diag d+2 row k+1; (k+1,j),(k,j+1) -> diag d+1
                    float diag_c  = (k + 1 <= m && j + 1 <= n) ? Ep2[k + 1] + C2[k + 1] : SDTW_INF;
                    float below_c = (k + 1 <= m)               ? Ep1[k + 1] + C1[k + 1] : SDTW_INF;
                    float right_c = (j + 1 <= n)               ? Ep1[k]     + C1[k]     : SDTW_INF;
                    e = softmin3(diag_c, below_c, right_c, inv_g, g);
                    Bd[base + k] = e;
                }
                if (k <= m + 1) Enw[k] = e;
            }
            // park the prefetched cost diagonal into LDS (wait lands after compute)
            #pragma unroll
            for (int s = 0; s < SLOTS; ++s) {
                int r = tid + s * DTW_THREADS;
                if (r <= m + 1) Cn[r] = cl[s];
            }
            __syncthreads();
            int t = pe2; pe2 = pe1; pe1 = pen; pen = t;
            int u = pc2; pc2 = pc1; pc1 = pcn; pcn = u;
        }
    }
}

// ---------------------------------------------------------------------------
// Fused alignment + statistics over diagonal-packed buffers (linear streams).
// stats: [0]=total, [1]=temporal, [2]=boundary, [3..18]=seg_sum, [19..34]=seg_jw
// ---------------------------------------------------------------------------
__global__ __launch_bounds__(256) void align_stats(
    const float* __restrict__ Fd, const float* __restrict__ Bd, const float* __restrict__ Cd,
    const int* __restrict__ off, const int* __restrict__ gb, int ngb,
    int M, int N, float g, float* __restrict__ stats) {

    __shared__ float sh[3 + 2 * MAX_SEG];
    int tid = threadIdx.x;
    if (tid < 3 + 2 * MAX_SEG) sh[tid] = 0.f;
    __syncthreads();

    int t = blockIdx.x * 256 + tid;
    if (t < M * N) {
        // binary search diagonal: off[d] <= t < off[d+1], d in [2, M+N]
        int lo = 2, hi = M + N;
        while (lo < hi) {
            int mid = (lo + hi + 1) >> 1;
            if (off[mid] <= t) lo = mid; else hi = mid - 1;
        }
        int d = lo;
        int ilo = imax(1, d - N);
        int i1 = ilo + (t - off[d]);   // 1-based
        int j1 = d - i1;
        int i = i1 - 1, j = j1 - 1;    // 0-based

        float dist = Fd[off[M + N]];
        float f = Fd[t], b = Bd[t], c = Cd[t];
        float lp = fminf(-(f + c + b - dist) / g, 0.0f);
        float a = __expf(lp);

        float denM = (float)(M > 1 ? M - 1 : 1);
        float denN = (float)(N > 1 ? N - 1 : 1);
        float ipos = (float)i / denM;
        float jpos = (float)j / denN;

        atomicAdd(&sh[0], a);
        float dt = ipos - jpos;
        atomicAdd(&sh[1], a * dt * dt);

        for (int q = 0; q < ngb; ++q) {
            int bv = gb[q];
            if (bv > 0 && bv < M && i == bv) {
                float dev = jpos - (float)bv / denM;
                atomicAdd(&sh[2], a * dev * dev);
            }
        }
        for (int s = 0; s + 1 < ngb && s < MAX_SEG; ++s) {
            int st = gb[s];
            int en = gb[s + 1]; if (en > M) en = M;
            if (i >= st && i < en) {
                atomicAdd(&sh[3 + s], a);
                atomicAdd(&sh[3 + MAX_SEG + s], a * jpos);
            }
        }
    }
    __syncthreads();
    if (tid < 3 + 2 * MAX_SEG) atomicAdd(&stats[tid], sh[tid]);
}

// ---------------------------------------------------------------------------
__global__ void finalize_loss(
    const float* __restrict__ stats, const float* __restrict__ Fd,
    const int* __restrict__ off, const int* __restrict__ gb, int ngb, int M, int N,
    float* __restrict__ out) {

    float dist = Fd[off[M + N]];
    float total = stats[0];
    float mn = (float)(M < N ? M : N);
    float scale = (total > 1e-8f) ? mn / fmaxf(total, 1e-8f) : 1.0f;

    float temporal = scale * stats[1];

    int nint = 0;
    for (int q = 0; q < ngb; ++q) if (gb[q] > 0 && gb[q] < M) nint++;
    float boundary = (nint > 0) ? scale * stats[2] / (float)nint : 0.0f;

    float pos[MAX_SEG];
    int np = 0;
    float order = 0.0f;
    if (ngb > 3) {
        for (int s = 0; s + 1 < ngb && s < MAX_SEG; ++s) {
            int st = gb[s], en = gb[s + 1];
            if (st >= M || en > M || st >= en) continue;
            float mass = fmaxf(scale * stats[3 + s], 1e-8f);
            pos[np++] = scale * stats[3 + MAX_SEG + s] / mass;
        }
        for (int k = 1; k < np; ++k) order += fmaxf(pos[k - 1] - pos[k], 0.0f);
        if (np > 1) order /= (float)(np - 1);
    }

    float cov = 0.0f;
    int nsteps = ngb - 1;
    if (nsteps > 0) {
        for (int s = 0; s < nsteps && s < MAX_SEG; ++s) {
            int st = gb[s];
            int en = gb[s + 1]; if (en > M) en = M;
            if (st >= en) continue;
            cov += fmaxf((float)(en - st) * 0.5f - scale * stats[3 + s], 0.0f);
        }
        cov /= (float)nsteps;
    }

    out[0] = 0.3f * dist + 0.2f * temporal + 0.2f * boundary
           + 0.15f * order + 0.15f * cov;
}

// ---------------------------------------------------------------------------
extern "C" void kernel_launch(void* const* d_in, const int* in_sizes, int n_in,
                              void* d_out, int out_size, void* d_ws, size_t ws_size,
                              hipStream_t stream) {
    const float* pred   = (const float*)d_in[0];   // (N, 64)
    const float* target = (const float*)d_in[1];   // (M, 64)
    const int*   gb     = (const int*)d_in[2];
    const int D   = 64;
    const int N   = in_sizes[0] / D;
    const int M   = in_sizes[1] / D;
    const int ngb = in_sizes[2];
    const float g = SDTW_GAMMA;

    char* ws = (char*)d_ws;
    _Float16* Tg16 = (_Float16*)ws;                              // M*64 halfs
    _Float16* Pd16 = (_Float16*)(ws + (size_t)M * D * 2);        // N*64 halfs
    float* t2    = (float*)(ws + (size_t)(M + N) * D * 2);       // M floats
    float* p2    = t2 + M;                                        // N floats
    float* stats = p2 + N;                                        // 64 floats
    int*   off   = (int*)(stats + 64);                            // m+n+2 ints
    float* Cd = (float*)(ws + (size_t)(1 << 20));                 // M*N packed
    float* Fd = Cd + (size_t)M * N;                               // M*N packed
    float* Bd = Fd + (size_t)M * N;                               // M*N packed

    zero_stats<<<1, 64, 0, stream>>>(stats);
    build_offsets<<<1, 1, 0, stream>>>(off, M, N);

    {
        int nT = M * D, nP = N * D;
        cvt_f32_to_f16<<<(nT + 255) / 256, 256, 0, stream>>>(target, Tg16, nT);
        cvt_f32_to_f16<<<(nP + 255) / 256, 256, 0, stream>>>(pred,   Pd16, nP);
    }
    row_norms<<<(M + 7) / 8, 256, 0, stream>>>(target, t2, M, D);
    row_norms<<<(N + 7) / 8, 256, 0, stream>>>(pred,   p2, N, D);

    {
        int tiles = (M / 16) * (N / 16);
        cost_wmma<<<(tiles + 7) / 8, 256, 0, stream>>>(Tg16, Pd16, t2, p2, off, Cd, M, N);
    }

    // forward (block 0) and backward (block 1) concurrently
    dtw_both<<<2, DTW_THREADS, 0, stream>>>(Cd, Fd, Bd, off, M, N, g);

    {
        int cells = M * N;
        int blocks = (cells + 255) / 256;
        align_stats<<<blocks, 256, 0, stream>>>(Fd, Bd, Cd, off, gb, ngb, M, N, g, stats);
    }
    finalize_loss<<<1, 1, 0, stream>>>(stats, Fd, off, gb, ngb, M, N, (float*)d_out);
}
